// AutoFIModel_15461882266238
// MI455X (gfx1250) — matrix operand order repaired and checked by reading, not compiled, not run on hardware
//
#include <hip/hip_runtime.h>
#include <hip/hip_bf16.h>

typedef __attribute__((ext_vector_type(16))) _Float16 v16h;
typedef __attribute__((ext_vector_type(8)))  _Float16 v8h;
typedef __attribute__((ext_vector_type(8)))  float    v8f;
typedef int v4i __attribute__((vector_size(16)));   // param type per hipcc diagnostic

#if __has_builtin(__builtin_amdgcn_global_load_async_to_lds_b128) && \
    __has_builtin(__builtin_amdgcn_s_wait_asynccnt)
#define HAVE_ASYNC_LDS 1
#endif

#define NUM_FIELDS 24
#define FIELD_DIM  2000
#define VOCAB      (NUM_FIELDS * FIELD_DIM)
#define EMBED_DIM  64
#define NUM_PAIRS  276
#define NUM_INTER  16
#define BATCH      4096

// ---------------- workspace layout (all 256B aligned) ----------------
constexpr size_t OFF_EMB32  = 0;                               // 4096*24*64 f32 = 25165824 B
constexpr size_t OFF_EMB16  = OFF_EMB32 + 25165824;            // 4096*24*64 f16 = 12582912 B
constexpr size_t OFF_LIN    = OFF_EMB16 + 12582912;            // 4096 f32       = 16384 B
constexpr size_t OFF_SCORES = OFF_LIN   + 16384;               // 276*4096 f32   = 4521984 B
constexpr size_t OFF_MEAN   = OFF_SCORES + 4521984;            // 276 f32 (pad 1280)
constexpr size_t OFF_SEL    = OFF_MEAN  + 1280;                // 48 ints (pad 256)
constexpr size_t OFF_W1B    = OFF_SEL   + 256;                 // 16 tiles * 512 f16 = 16384 B
constexpr size_t OFF_W2B    = OFF_W1B   + 16384;               // 4 tiles * 512 f16  = 4096 B

// ---------------- 0) pre-swizzle weights into WMMA B-register layout ----------------
// B tile (32x16 f16 operand): lane L holds K = kb..kb+15 (kb = L<16?0:16) at column
// n = L&15, as 16 contiguous halfs. Tile-major: [tile][lane][16].
__global__ __launch_bounds__(256)
void prep_kernel(const float* __restrict__ w1, const float* __restrict__ w2,
                 _Float16* __restrict__ w1B, _Float16* __restrict__ w2B) {
    const int tid = threadIdx.x;
    for (int d = tid; d < 16 * 512; d += 256) {          // W1: (128x64) -> 4 kc x 4 nt tiles
        const int tile = d >> 9, within = d & 511;
        const int lane = within >> 4, t = within & 15;
        const int kc = tile >> 2, nt = tile & 3;
        const int kb = (lane < 16) ? 0 : 16, n = lane & 15;
        w1B[d] = (_Float16)w1[(kc * 32 + kb + t) * 64 + nt * 16 + n];
    }
    for (int d = tid; d < 4 * 512; d += 256) {           // W2: (64x32) -> 2 kc x 2 nt tiles
        const int tile = d >> 9, within = d & 511;
        const int lane = within >> 4, t = within & 15;
        const int kc = tile >> 1, nt = tile & 1;
        const int kb = (lane < 16) ? 0 : 16, n = lane & 15;
        w2B[d] = (_Float16)w2[(kc * 32 + kb + t) * 32 + nt * 16 + n];
    }
}

// ---------------- 1) gather embeddings + linear part ----------------
__global__ __launch_bounds__(64)
void gather_kernel(const int* __restrict__ x, const float* __restrict__ tables,
                   const float* __restrict__ fc_w, const float* __restrict__ fc_b,
                   float* __restrict__ emb32, _Float16* __restrict__ emb16,
                   float* __restrict__ lin) {
    const int b = blockIdx.x;
    const int d = threadIdx.x;
    float linacc = fc_b[0];
    for (int f = 0; f < NUM_FIELDS; ++f) {
        const int idx = x[b * NUM_FIELDS + f] + f * FIELD_DIM;
        const float v = tables[((size_t)f * VOCAB + (size_t)idx) * EMBED_DIM + d];
        const size_t o = ((size_t)(b * NUM_FIELDS + f) << 6) + d;
        emb32[o] = v;
        emb16[o] = (_Float16)v;
        if (d == 0) linacc += fc_w[idx];
    }
    if (d == 0) lin[b] = linacc;
}

// ---------------- 2) pair MLP via WMMA ----------------
// grid: (BATCH/128, NUM_PAIRS), block: 256 (8 waves). Each wave: 16 batch rows.
__global__ __launch_bounds__(256)
void mlp_kernel(const _Float16* __restrict__ w1B, const float* __restrict__ b1,
                const _Float16* __restrict__ w2B, const float* __restrict__ b2,
                const float* __restrict__ w3, const float* __restrict__ b3,
                const _Float16* __restrict__ emb16, float* __restrict__ scores) {
    __shared__ __align__(32) _Float16 sW1B[16 * 512];
    __shared__ __align__(32) _Float16 sW2B[4 * 512];
    __shared__ __align__(32) _Float16 stage[8][16 * 64];

    const int tid = threadIdx.x;

    // stage pre-swizzled weights into LDS via CDNA5 async-to-LDS engine
#ifdef HAVE_ASYNC_LDS
    {
        const char* g1 = (const char*)w1B;
        char*       l1 = (char*)sW1B;
#pragma unroll
        for (int it = 0; it < 4; ++it) {
            const int off = (it * 256 + tid) * 16;       // 16 KB in 4 x 4KB sweeps
            __builtin_amdgcn_global_load_async_to_lds_b128(
                (v4i*)(g1 + off), (v4i*)(l1 + off), 0, 0);
        }
        __builtin_amdgcn_global_load_async_to_lds_b128(
            (v4i*)((const char*)w2B + tid * 16),
            (v4i*)((char*)sW2B + tid * 16), 0, 0);
        __builtin_amdgcn_s_wait_asynccnt(0);
    }
#else
    for (int t = tid; t < 16 * 512; t += 256) sW1B[t] = w1B[t];
    for (int t = tid; t < 4 * 512;  t += 256) sW2B[t] = w2B[t];
#endif
    __syncthreads();

    const int lane  = tid & 31;
    const int wv    = tid >> 5;
    const int n     = lane & 15;
    const int koff  = (lane < 16) ? 0 : 16;   // A-layout K offset per lane half
    const int mBase = (lane < 16) ? 0 : 8;    // C-layout M base per lane half

    // decode pair p -> (fi, fj) matching triu_indices(24, k=1) row-major order
    const int p = blockIdx.y;
    int fi = 0, rem = p, cnt = NUM_FIELDS - 1;
    while (rem >= cnt) { rem -= cnt; ++fi; --cnt; }
    const int fj = fi + 1 + rem;

    const int rowBase = blockIdx.x * 128 + wv * 16;
    const int arow    = rowBase + n;          // batch row this lane supplies for A

    // ---- load A (16x128 f16) in WMMA A layout: 4 K-chunks of 32 ----
    v16h A[4];
#pragma unroll
    for (int kc = 0; kc < 4; ++kc) {
        const int field = (kc < 2) ? fi : fj;
        const int dbase = (kc & 1) * 32 + koff;
        const _Float16* src = emb16 + (((size_t)(arow * NUM_FIELDS + field)) << 6);
        v8h lo = *(const v8h*)(src + dbase);
        v8h hi = *(const v8h*)(src + dbase + 8);
        A[kc] = __builtin_shufflevector(lo, hi, 0, 1, 2, 3, 4, 5, 6, 7,
                                        8, 9, 10, 11, 12, 13, 14, 15);
    }

    // ---- layer 1: C1(16x64) = A(16x128) @ W1(128x64) ----
    v8f C1[4];
#pragma unroll
    for (int nt = 0; nt < 4; ++nt) {
        v8f acc = {};
#pragma unroll
        for (int kc = 0; kc < 4; ++kc) {
            const v16h B = *(const v16h*)(sW1B + (((kc << 2) + nt) << 9) + (lane << 4));
            acc = __builtin_amdgcn_wmma_f32_16x16x32_f16(false, A[kc], false, B,
                                                         (short)0, acc, false, false);
        }
        C1[nt] = acc;
    }

    // bias + relu; re-layout C->A via per-wave LDS stage (f16)
    _Float16* st = &stage[wv][0];
#pragma unroll
    for (int nt = 0; nt < 4; ++nt) {
        const float bb = b1[nt * 16 + n];
#pragma unroll
        for (int v = 0; v < 8; ++v) {
            float val = C1[nt][v] + bb;
            val = val > 0.f ? val : 0.f;
            st[(mBase + v) * 64 + nt * 16 + n] = (_Float16)val;
        }
    }
    __syncthreads();

    // ---- layer 2: C2(16x32) = H1(16x64) @ W2(64x32) ----
    v16h A2[2];
#pragma unroll
    for (int kc = 0; kc < 2; ++kc) {
        const _Float16* s2 = st + n * 64 + kc * 32;
        v8h lo = *(const v8h*)(s2 + koff);
        v8h hi = *(const v8h*)(s2 + 8 + koff);
        A2[kc] = __builtin_shufflevector(lo, hi, 0, 1, 2, 3, 4, 5, 6, 7,
                                         8, 9, 10, 11, 12, 13, 14, 15);
    }

    v8f C2[2];
#pragma unroll
    for (int nt = 0; nt < 2; ++nt) {
        v8f acc = {};
#pragma unroll
        for (int kc = 0; kc < 2; ++kc) {
            const v16h B = *(const v16h*)(sW2B + (((kc << 1) + nt) << 9) + (lane << 4));
            acc = __builtin_amdgcn_wmma_f32_16x16x32_f16(false, A2[kc], false, B,
                                                         (short)0, acc, false, false);
        }
        C2[nt] = acc;
    }

    // ---- head: scores = relu(C2 + b2) @ w3 + b3, cross-lane reduce over N ----
    const float w3a = w3[n],      w3b = w3[16 + n];
    const float b2a = b2[n],      b2b = b2[16 + n];
    const float b3v = b3[0];
#pragma unroll
    for (int v = 0; v < 8; ++v) {
        float h0 = C2[0][v] + b2a; h0 = h0 > 0.f ? h0 : 0.f;
        float h1 = C2[1][v] + b2b; h1 = h1 > 0.f ? h1 : 0.f;
        float pr = h0 * w3a + h1 * w3b;
        pr += __shfl_xor(pr, 1, 32);
        pr += __shfl_xor(pr, 2, 32);
        pr += __shfl_xor(pr, 4, 32);
        pr += __shfl_xor(pr, 8, 32);
        if (n == 0)
            scores[(size_t)p * BATCH + rowBase + mBase + v] = pr + b3v;
    }
}

// ---------------- 3) batch-mean of scores per pair ----------------
__global__ __launch_bounds__(256)
void mean_kernel(const float* __restrict__ scores, float* __restrict__ mean) {
    __shared__ float red[256];
    const int p = blockIdx.x;
    float s = 0.f;
    for (int t = threadIdx.x; t < BATCH; t += 256) s += scores[(size_t)p * BATCH + t];
    red[threadIdx.x] = s;
    __syncthreads();
    for (int off = 128; off > 0; off >>= 1) {
        if (threadIdx.x < off) red[threadIdx.x] += red[threadIdx.x + off];
        __syncthreads();
    }
    if (threadIdx.x == 0) mean[p] = red[0] * (1.f / BATCH);
}

// ---------------- 4) exact serial top-16 (descending, lowest-index ties) ----------------
__global__ void topk_kernel(const float* __restrict__ mean, int* __restrict__ sel) {
    if (threadIdx.x != 0 || blockIdx.x != 0) return;
    float m[NUM_PAIRS];
    for (int q = 0; q < NUM_PAIRS; ++q) m[q] = mean[q];
    for (int t = 0; t < NUM_INTER; ++t) {
        int best = 0; float bv = -3.4e38f;
        for (int q = 0; q < NUM_PAIRS; ++q)
            if (m[q] > bv) { bv = m[q]; best = q; }
        m[best] = -3.4e38f;
        int fi = 0, rem = best, cnt = NUM_FIELDS - 1;
        while (rem >= cnt) { rem -= cnt; ++fi; --cnt; }
        sel[t] = best; sel[16 + t] = fi; sel[32 + t] = fi + 1 + rem;
    }
}

// ---------------- 5) selected crossed products + projection + combine ----------------
__global__ __launch_bounds__(64)
void final_kernel(const float* __restrict__ emb32, const float* __restrict__ scores,
                  const int* __restrict__ sel, const float* __restrict__ lin,
                  const float* __restrict__ proj_w, const float* __restrict__ proj_b,
                  const float* __restrict__ bias, float* __restrict__ out) {
    __shared__ float red[64];
    const int b = blockIdx.x, d = threadIdx.x;
    float s = 0.f;
    for (int t = 0; t < NUM_INTER; ++t) {
        const int p = sel[t], fi = sel[16 + t], fj = sel[32 + t];
        const float sc = scores[(size_t)p * BATCH + b];
        const float e1 = emb32[(((size_t)(b * NUM_FIELDS + fi)) << 6) + d];
        const float e2 = emb32[(((size_t)(b * NUM_FIELDS + fj)) << 6) + d];
        s += e1 * e2 * sc;
    }
    red[d] = s * proj_w[d];
    __syncthreads();
    for (int off = 32; off > 0; off >>= 1) {
        if (d < off) red[d] += red[d + off];
        __syncthreads();
    }
    if (d == 0) out[b] = lin[b] + red[0] + proj_b[0] + bias[0];
}

// ---------------- launch ----------------
extern "C" void kernel_launch(void* const* d_in, const int* in_sizes, int n_in,
                              void* d_out, int out_size, void* d_ws, size_t ws_size,
                              hipStream_t stream) {
    (void)in_sizes; (void)n_in; (void)out_size; (void)ws_size;
    const int*   x      = (const int*)d_in[0];
    const float* tables = (const float*)d_in[1];
    const float* fc_w   = (const float*)d_in[2];
    const float* fc_b   = (const float*)d_in[3];
    const float* w1     = (const float*)d_in[4];
    const float* b1     = (const float*)d_in[5];
    const float* w2     = (const float*)d_in[6];
    const float* b2     = (const float*)d_in[7];
    const float* w3     = (const float*)d_in[8];
    const float* b3     = (const float*)d_in[9];
    const float* proj_w = (const float*)d_in[10];
    const float* proj_b = (const float*)d_in[11];
    const float* bias   = (const float*)d_in[12];
    float* out = (float*)d_out;

    char* ws = (char*)d_ws;
    float*    emb32  = (float*)(ws + OFF_EMB32);
    _Float16* emb16  = (_Float16*)(ws + OFF_EMB16);
    float*    lin    = (float*)(ws + OFF_LIN);
    float*    scores = (float*)(ws + OFF_SCORES);
    float*    mean   = (float*)(ws + OFF_MEAN);
    int*      sel    = (int*)(ws + OFF_SEL);
    _Float16* w1B    = (_Float16*)(ws + OFF_W1B);
    _Float16* w2B    = (_Float16*)(ws + OFF_W2B);

    prep_kernel<<<1, 256, 0, stream>>>(w1, w2, w1B, w2B);
    gather_kernel<<<BATCH, 64, 0, stream>>>(x, tables, fc_w, fc_b, emb32, emb16, lin);

    dim3 g2(BATCH / 128, NUM_PAIRS);
    mlp_kernel<<<g2, 256, 0, stream>>>(w1B, b1, w2B, b2, w3, b3, emb16, scores);

    mean_kernel<<<NUM_PAIRS, 256, 0, stream>>>(scores, mean);
    topk_kernel<<<1, 32, 0, stream>>>(mean, sel);
    final_kernel<<<BATCH, 64, 0, stream>>>(emb32, scores, sel, lin,
                                           proj_w, proj_b, bias, out);
}
